// WaveFunctionNormal_43911745634895
// MI455X (gfx1250) — matrix operand order repaired and checked by reading, not compiled, not run on hardware
//
#include <hip/hip_runtime.h>
#include <hip/hip_bf16.h>
#include <math.h>

#define SITES 64
#define HD    256      // H0 == H1 == 256
#define BATCH 2048
#define TB    64       // batch rows per block
#define LDX   72       // LDS stride for x tile   (64 + 8, rows 16B-aligned)
#define LDH   264      // LDS stride for h1 tile  (256 + 8, rows 16B-aligned)
#define LDK   40       // LDS stride for weight slab columns (32 + 8; 80B => 16B-aligned)

#define W1E (SITES * SITES * HD)   // 1,048,576
#define W2E (SITES * HD * HD)      // 4,194,304

typedef __attribute__((ext_vector_type(16))) __bf16 v16bf;
typedef __attribute__((ext_vector_type(8)))  __bf16 v8bf;
typedef __attribute__((ext_vector_type(4)))  __bf16 v4bf;
typedef __attribute__((ext_vector_type(8)))  float  v8f;

__device__ __forceinline__ float silu_f(float z) {
    return __fdividef(z, 1.0f + __expf(-z));   // fast rcp path, no v_div_scale chain
}

// A-fragment (16x32 bf16) from LDS, row-major [rows][ld].
// Lane L holds row M = L&15; lanes 0-15 carry K 0-7 & 16-23,
// lanes 16-31 carry K 8-15 & 24-31  -> two contiguous 16B DS loads.
__device__ __forceinline__ v16bf load_a_frag(const __bf16* sm, int row, int ld,
                                             int kbase, int lane) {
    const int koff = (lane >> 4) << 3;                 // 0 or 8
    const __bf16* p = sm + row * ld + kbase;
    v8bf lo = *(const v8bf*)(p + koff);
    v8bf hi = *(const v8bf*)(p + 16 + koff);
    v16bf a;
#pragma unroll
    for (int i = 0; i < 8; ++i) { a[i] = lo[i]; a[i + 8] = hi[i]; }
    return a;
}

// B-fragment (32x16 bf16) from the transposed LDS weight slab s_w[N][K'].
// Lane L: N = nt*16 + (L&15); K = (L>>4)*8 + 0..7 and 16 + (L>>4)*8 + 0..7.
__device__ __forceinline__ v16bf load_b_frag_lds(const __bf16* s_w, int nt, int lane) {
    const int col = nt * 16 + (lane & 15);
    const int k0  = (lane >> 4) << 3;                  // 0 or 8
    const __bf16* p = s_w + col * LDK + k0;
    v8bf lo = *(const v8bf*)(p);
    v8bf hi = *(const v8bf*)(p + 16);
    v16bf b;
#pragma unroll
    for (int i = 0; i < 8; ++i) { b[i] = lo[i]; b[i + 8] = hi[i]; }
    return b;
}

// 8 WMMAs sharing one A fragment, with B fragments double-buffered so the
// next tile's ds_load_b128 pair is in flight under the current WMMA.
__device__ __forceinline__ void wmma_row(const __bf16* s_w, const v16bf a,
                                         v8f acc[8], int ng, int lane) {
    v16bf bcur = load_b_frag_lds(s_w, ng + 0, lane);
#pragma unroll
    for (int j = 0; j < 8; ++j) {
        const v16bf bnxt = (j < 7) ? load_b_frag_lds(s_w, ng + j + 1, lane) : bcur;
        acc[j] = __builtin_amdgcn_wmma_f32_16x16x32_bf16(false, a, false, bcur,
                                                         (short)0, acc[j], false, false);
        bcur = bnxt;
    }
}

// Stage one 32(K) x 256(N) weight slab into LDS transposed: s_w[N*LDK + K].
// PRE=true  : copy from pre-converted bf16 weights in workspace.
// PRE=false : convert fp32 -> bf16 on the fly.
template <bool PRE>
__device__ __forceinline__ void stage_w_slab(__bf16* s_w,
                                             const float* __restrict__ wf,
                                             const __bf16* __restrict__ wb,
                                             int kbase, int tid) {
#pragma unroll 1
    for (int i = tid; i < 32 * 64; i += 256) {         // each i covers 4 N values
        const int K  = i >> 6;
        const int N4 = (i & 63) << 2;
        const int g  = (kbase + K) * HD + N4;
        __bf16 e0, e1, e2, e3;
        if (PRE) {
            v4bf v = *(const v4bf*)&wb[g];
            e0 = v[0]; e1 = v[1]; e2 = v[2]; e3 = v[3];
        } else {
            float4 v = *(const float4*)&wf[g];
            e0 = (__bf16)v.x; e1 = (__bf16)v.y; e2 = (__bf16)v.z; e3 = (__bf16)v.w;
        }
        s_w[(N4 + 0) * LDK + K] = e0;
        s_w[(N4 + 1) * LDK + K] = e1;
        s_w[(N4 + 2) * LDK + K] = e2;
        s_w[(N4 + 3) * LDK + K] = e3;
    }
}

template <bool PRE>
__global__ __launch_bounds__(256)
void nqs_mlp_site_kernel(const int* __restrict__ x,
                         const float* __restrict__ aw1, const float* __restrict__ ab1,
                         const float* __restrict__ aw2, const float* __restrict__ ab2,
                         const float* __restrict__ aw3, const float* __restrict__ ab3,
                         const float* __restrict__ pw1, const float* __restrict__ pb1,
                         const float* __restrict__ pw2, const float* __restrict__ pb2,
                         const float* __restrict__ pw3, const float* __restrict__ pb3,
                         const __bf16* __restrict__ aw1b, const __bf16* __restrict__ aw2b,
                         const __bf16* __restrict__ pw1b, const __bf16* __restrict__ pw2b,
                         float* __restrict__ amp_acc, float* __restrict__ pha_acc) {
    __shared__ __bf16 s_x[TB * LDX];       // masked inputs, bf16        (9.2 KB)
    __shared__ __bf16 s_h1[TB * LDH];      // layer-1 activations, bf16  (33.8 KB)
    __shared__ __bf16 s_w[HD * LDK];       // transposed weight slab     (20.5 KB)
    __shared__ float  s_out[TB][2];        // layer-3 accumulators       (0.5 KB)

    const int tid  = threadIdx.x;
    const int lane = tid & 31;             // wave32
    const int wid  = tid >> 5;             // 8 waves
    const int b0   = blockIdx.x * TB;      // batch tile base
    const int s    = blockIdx.y;           // site
    const int mlp  = blockIdx.z;           // 0 = amplitude, 1 = phase

    const float* b1 = mlp ? pb1 : ab1;
    const float* b2 = mlp ? pb2 : ab2;
    const float* b3 = mlp ? pb3 : ab3;
    const float* w1s = (mlp ? pw1 : aw1) + (size_t)s * SITES * HD;
    const float* w2s = (mlp ? pw2 : aw2) + (size_t)s * HD * HD;
    const float* w3s = (mlp ? pw3 : aw3) + (size_t)s * HD * 2;
    const __bf16* w1bs = PRE ? ((mlp ? pw1b : aw1b) + (size_t)s * SITES * HD) : nullptr;
    const __bf16* w2bs = PRE ? ((mlp ? pw2b : aw2b) + (size_t)s * HD * HD)   : nullptr;

    // ---- stage masked activations: xm[b, j] = x[b, j] if j < s else 0 ----
    for (int i = tid; i < TB * SITES; i += 256) {
        const int row = i >> 6, col = i & 63;
        const float v = (col < s) ? (float)x[(b0 + row) * SITES + col] : 0.0f;
        s_x[row * LDX + col] = (__bf16)v;
    }
    if (tid < TB * 2) ((float*)s_out)[tid] = 0.0f;
    __syncthreads();

    const int mt   = wid & 3;              // m-tile (4 x 16 rows = 64 rows)
    const int ng   = (wid >> 2) * 8;       // first of 8 owned n-tiles
    const int arow = mt * 16 + (lane & 15);

    // ================= layer 1: [64,64] x [64,256], K = 2 slabs of 32 =========
    v8f acc[8] = {};
#pragma unroll 1
    for (int kc = 0; kc < 2; ++kc) {
        stage_w_slab<PRE>(s_w, w1s, w1bs, kc * 32, tid);
        __syncthreads();
        const v16bf a = load_a_frag(s_x, arow, LDX, kc * 32, lane);
        wmma_row(s_w, a, acc, ng, lane);
        __syncthreads();
    }
#pragma unroll
    for (int j = 0; j < 8; ++j) {
        const int col  = (ng + j) * 16 + (lane & 15);
        const float bi = b1[s * HD + col];
#pragma unroll
        for (int r = 0; r < 8; ++r) {
            const int row = mt * 16 + r + ((lane >> 4) << 3);
            s_h1[row * LDH + col] = (__bf16)silu_f(acc[j][r] + bi);
        }
    }
    __syncthreads();

    // ========== layer 2 (+fused layer 3): [64,256] x [256,256], 8 slabs ========
#pragma unroll
    for (int j = 0; j < 8; ++j) acc[j] = (v8f){};
#pragma unroll 1
    for (int kc = 0; kc < 8; ++kc) {
        stage_w_slab<PRE>(s_w, w2s, w2bs, kc * 32, tid);
        __syncthreads();
        const v16bf a = load_a_frag(s_h1, arow, LDH, kc * 32, lane);
        wmma_row(s_w, a, acc, ng, lane);
        __syncthreads();
    }
    // fully unrolled epilogue: keeps acc[] in VGPRs (no dynamic indexing -> no spills)
#pragma unroll
    for (int j = 0; j < 8; ++j) {
        const int col   = (ng + j) * 16 + (lane & 15);
        const float bi  = b2[s * HD + col];
        const float w30 = w3s[col * 2 + 0];
        const float w31 = w3s[col * 2 + 1];
#pragma unroll
        for (int r = 0; r < 8; ++r) {
            const float h = silu_f(acc[j][r] + bi);
            float p0 = h * w30, p1 = h * w31;
#pragma unroll
            for (int m = 8; m >= 1; m >>= 1) {        // reduce across 16 N-lanes
                p0 += __shfl_xor(p0, m, 32);
                p1 += __shfl_xor(p1, m, 32);
            }
            if ((lane & 15) == 0) {
                const int row = mt * 16 + r + ((lane >> 4) << 3);
                atomicAdd(&s_out[row][0], p0);
                atomicAdd(&s_out[row][1], p1);
            }
        }
    }
    __syncthreads();

    // ---- normalize (amp), select by spin, accumulate per-batch sums ----
    if (tid < TB) {
        const int gb  = b0 + tid;
        const int sel = x[gb * SITES + s];
        const float z0 = s_out[tid][0] + b3[s * 2 + 0];
        const float z1 = s_out[tid][1] + b3[s * 2 + 1];
        const float zs = sel ? z1 : z0;
        if (mlp == 0) {
            const float m   = 2.0f * fmaxf(z0, z1);
            const float lse = m + __logf(__expf(2.0f * z0 - m) + __expf(2.0f * z1 - m));
            atomicAdd(&amp_acc[gb], zs - 0.5f * lse);
        } else {
            atomicAdd(&pha_acc[gb], zs);
        }
    }
}

__global__ void nqs_cvt_bf16_kernel(const float* __restrict__ src,
                                    __bf16* __restrict__ dst, int n4) {
    const int i = blockIdx.x * blockDim.x + threadIdx.x;
    if (i < n4) {
        const float4 v = ((const float4*)src)[i];
        v4bf o;
        o[0] = (__bf16)v.x; o[1] = (__bf16)v.y; o[2] = (__bf16)v.z; o[3] = (__bf16)v.w;
        *(v4bf*)&dst[i * 4] = o;
    }
}

__global__ void nqs_zero_kernel(float* __restrict__ p, int n) {
    const int i = blockIdx.x * blockDim.x + threadIdx.x;
    if (i < n) p[i] = 0.0f;
}

__global__ void nqs_finalize_kernel(const float* __restrict__ amp_acc,
                                    const float* __restrict__ pha_acc,
                                    float* __restrict__ out) {
    const int b = blockIdx.x * blockDim.x + threadIdx.x;
    if (b < BATCH) {
        const float r = __expf(amp_acc[b]);
        const float p = pha_acc[b];
        out[2 * b + 0] = r * __cosf(p);   // complex64 interleaved
        out[2 * b + 1] = r * __sinf(p);
    }
}

extern "C" void kernel_launch(void* const* d_in, const int* in_sizes, int n_in,
                              void* d_out, int out_size, void* d_ws, size_t ws_size,
                              hipStream_t stream) {
    const int*   x   = (const int*)d_in[0];
    const float* aw1 = (const float*)d_in[1];  const float* ab1 = (const float*)d_in[2];
    const float* aw2 = (const float*)d_in[3];  const float* ab2 = (const float*)d_in[4];
    const float* aw3 = (const float*)d_in[5];  const float* ab3 = (const float*)d_in[6];
    const float* pw1 = (const float*)d_in[7];  const float* pb1 = (const float*)d_in[8];
    const float* pw2 = (const float*)d_in[9];  const float* pb2 = (const float*)d_in[10];
    const float* pw3 = (const float*)d_in[11]; const float* pb3 = (const float*)d_in[12];

    float* amp_acc = (float*)d_ws;
    float* pha_acc = amp_acc + BATCH;
    __bf16* wbuf   = (__bf16*)(pha_acc + BATCH);
    __bf16* aw1b = wbuf;
    __bf16* pw1b = aw1b + W1E;
    __bf16* aw2b = pw1b + W1E;
    __bf16* pw2b = aw2b + W2E;

    const size_t need = (size_t)2 * BATCH * sizeof(float)
                      + (size_t)2 * (W1E + W2E) * sizeof(__bf16);
    const bool pre = ws_size >= need;

    nqs_zero_kernel<<<(2 * BATCH + 255) / 256, 256, 0, stream>>>(amp_acc, 2 * BATCH);

    dim3 grid(BATCH / TB, SITES, 2);
    if (pre) {
        nqs_cvt_bf16_kernel<<<(W1E / 4 + 255) / 256, 256, 0, stream>>>(aw1, aw1b, W1E / 4);
        nqs_cvt_bf16_kernel<<<(W1E / 4 + 255) / 256, 256, 0, stream>>>(pw1, pw1b, W1E / 4);
        nqs_cvt_bf16_kernel<<<(W2E / 4 + 255) / 256, 256, 0, stream>>>(aw2, aw2b, W2E / 4);
        nqs_cvt_bf16_kernel<<<(W2E / 4 + 255) / 256, 256, 0, stream>>>(pw2, pw2b, W2E / 4);
        nqs_mlp_site_kernel<true><<<grid, 256, 0, stream>>>(x,
            aw1, ab1, aw2, ab2, aw3, ab3, pw1, pb1, pw2, pb2, pw3, pb3,
            aw1b, aw2b, pw1b, pw2b, amp_acc, pha_acc);
    } else {
        nqs_mlp_site_kernel<false><<<grid, 256, 0, stream>>>(x,
            aw1, ab1, aw2, ab2, aw3, ab3, pw1, pb1, pw2, pb2, pw3, pb3,
            nullptr, nullptr, nullptr, nullptr, amp_acc, pha_acc);
    }

    nqs_finalize_kernel<<<(BATCH + 255) / 256, 256, 0, stream>>>(amp_acc, pha_acc,
                                                                 (float*)d_out);
}
